// MultiHeadSelfAttention_7782480740945
// MI455X (gfx1250) — compile-verified, matrix operand order checked by reading, not run on hardware
//
#include <hip/hip_runtime.h>

// ---------------------------------------------------------------------------
// MHSA forward for MI455X (gfx1250, wave32, WMMA f16 -> f32 accum, TDM tiles)
// B=4, S=2048, E=1024, H=16, D=64, causal mask.
// ---------------------------------------------------------------------------

typedef __attribute__((ext_vector_type(16))) _Float16 v16h;
typedef __attribute__((ext_vector_type(8)))  float    v8f;
typedef unsigned int u32x4 __attribute__((ext_vector_type(4)));
typedef int          i32x4 __attribute__((ext_vector_type(4)));
typedef int          i32x8 __attribute__((ext_vector_type(8)));

union FragU {
  v16h     v;
  _Float16 h[16];
  uint32_t u[8];
};

#define WMMA_F16(a, b, c) \
  __builtin_amdgcn_wmma_f32_16x16x32_f16(false, (a), false, (b), (short)0, (c), false, false)

static constexpr int Bc = 4, Sc = 2048, Ec = 1024, Hc = 16, Dc = 64;
static constexpr int E3 = 3 * Ec;  // 3072

// ---------------------------------------------------------------------------
// fp32 -> f16 conversion
// ---------------------------------------------------------------------------
__global__ __launch_bounds__(256)
void cvt_f32_f16(const float* __restrict__ in, _Float16* __restrict__ out, int n) {
  int i = blockIdx.x * 256 + threadIdx.x;
  if (i < n) out[i] = (_Float16)in[i];
}

// ---------------------------------------------------------------------------
// Tensor Data Mover: 2-D f16 tile (tile_d0 elems/row x tile_d1 rows) from
// global (row pitch stride0 elems) into LDS at byte offset lds_off, rows
// stored contiguously. D# packing per CDNA5 ISA 8.3/8.4:
//   g0: count=1 | lds_addr | global_addr[56:0] | type=2
//   g1: data_size=1(2B) | tensor_dim0/1 | tile_dim0/1 (tile_dim2=0 -> 2D)
//       | tensor_dim0_stride
// ---------------------------------------------------------------------------
__device__ __forceinline__ void tdm_load_2d(uint32_t lds_off, const void* gptr,
                                            uint32_t td0, uint32_t td1,
                                            uint64_t stride0,
                                            uint32_t tile_d0, uint32_t tile_d1) {
  const uint64_t ga = (uint64_t)(uintptr_t)gptr;
  u32x4 g0;
  g0[0] = 1u;                                              // count=1, user desc
  g0[1] = lds_off;                                         // lds_addr
  g0[2] = (uint32_t)ga;                                    // global_addr[31:0]
  g0[3] = ((uint32_t)(ga >> 32) & 0x01FFFFFFu) | (2u << 30);  // [56:32] | type=2
  i32x8 g1;
  g1[0] = (int)0x00010000u;                                // data_size = 2 bytes
  g1[1] = (int)((td0 & 0xFFFFu) << 16);                    // tensor_dim0 lo
  g1[2] = (int)((td0 >> 16) | ((td1 & 0xFFFFu) << 16));    // dim0 hi | dim1 lo
  g1[3] = (int)((td1 >> 16) | (tile_d0 << 16));            // dim1 hi | tile_dim0
  g1[4] = (int)(tile_d1 & 0xFFFFu);                        // tile_dim1, tile_dim2=0
  g1[5] = (int)(uint32_t)stride0;                          // dim0_stride lo
  g1[6] = (int)((uint32_t)(stride0 >> 32) & 0xFFFFu);      // dim0_stride hi
  g1[7] = 0;
  i32x4 z4 = {0, 0, 0, 0};
#if defined(__clang_major__) && __clang_major__ >= 23
  i32x8 z8 = {0, 0, 0, 0, 0, 0, 0, 0};
  __builtin_amdgcn_tensor_load_to_lds(g0, g1, z4, z4, z8, 0);
#else
  __builtin_amdgcn_tensor_load_to_lds(g0, g1, z4, z4, 0);
#endif
}

// 16x32 A-fragment from LDS tile stored [rows][32] halves (K contiguous).
__device__ __forceinline__ v16h lds_afrag(const _Float16* base, int row, int lane) {
  FragU r;
  const uint32_t* p = (const uint32_t*)(base + (size_t)(row + (lane & 15)) * 32);
  const int h4 = (lane >> 4) << 2;
#pragma unroll
  for (int j = 0; j < 4; ++j) {
    r.u[j]     = p[h4 + j];
    r.u[4 + j] = p[8 + h4 + j];
  }
  return r.v;
}

// 32x16 B-fragment from LDS tile stored [k][128] halves (N contiguous).
__device__ __forceinline__ v16h lds_bfrag(const _Float16* base, int col, int lane) {
  FragU r;
  const _Float16* p = base + col + (lane & 15);
  const int kb = (lane >> 4) << 3;
#pragma unroll
  for (int j = 0; j < 8; ++j) {
    r.h[j]     = p[(size_t)(kb + j) * 128];
    r.h[8 + j] = p[(size_t)(16 + kb + j) * 128];
  }
  return r.v;
}

// ---------------------------------------------------------------------------
// 128x128-tile GEMM, BK=32, 256 threads = 8 waves (2x4), TDM double-buffered.
// C[M,N] = A[M,K] @ B[K,N] + bias ; A,B f16 row-major; C f16 or f32.
// ---------------------------------------------------------------------------
template <bool OUT_F16>
__global__ __launch_bounds__(256)
void gemm_wmma_tdm(const _Float16* __restrict__ A, const _Float16* __restrict__ Bm,
                   const float* __restrict__ bias, void* __restrict__ C,
                   int M, int N, int K) {
  __shared__ _Float16 As[2][128 * 32];   // [m][k]
  __shared__ _Float16 Bs[2][32 * 128];   // [k][n]
  const int tid  = threadIdx.x;
  const int lane = tid & 31;
  const int wid  = tid >> 5;
  const int wm   = wid >> 2;             // 0..1 -> 64 rows
  const int wn   = wid & 3;              // 0..3 -> 32 cols
  const int row0 = blockIdx.y * 128;
  const int col0 = blockIdx.x * 128;
  const int nIter = K >> 5;

  v8f acc[4][2] = {};

  if (wid == 0) {
    tdm_load_2d((uint32_t)(uintptr_t)&As[0][0], A + (size_t)row0 * K,
                (uint32_t)K, (uint32_t)M, (uint64_t)K, 32u, 128u);
    tdm_load_2d((uint32_t)(uintptr_t)&Bs[0][0], Bm + (size_t)col0,
                (uint32_t)N, (uint32_t)K, (uint64_t)N, 128u, 32u);
  }

  for (int it = 0; it < nIter; ++it) {
    const int buf = it & 1;
    if (wid == 0) {
      if (it + 1 < nIter) {
        const int k1 = (it + 1) << 5;
        tdm_load_2d((uint32_t)(uintptr_t)&As[buf ^ 1][0],
                    A + (size_t)row0 * K + k1,
                    (uint32_t)K, (uint32_t)M, (uint64_t)K, 32u, 128u);
        tdm_load_2d((uint32_t)(uintptr_t)&Bs[buf ^ 1][0],
                    Bm + (size_t)k1 * N + col0,
                    (uint32_t)N, (uint32_t)K, (uint64_t)N, 128u, 32u);
        __builtin_amdgcn_s_wait_tensorcnt(2);   // current buf's pair done
      } else {
        __builtin_amdgcn_s_wait_tensorcnt(0);
      }
    }
    __syncthreads();                            // buf ready for all waves

    v16h bf[2];
#pragma unroll
    for (int nt = 0; nt < 2; ++nt)
      bf[nt] = lds_bfrag(Bs[buf], wn * 32 + nt * 16, lane);
#pragma unroll
    for (int mt = 0; mt < 4; ++mt) {
      v16h af = lds_afrag(As[buf], wm * 64 + mt * 16, lane);
#pragma unroll
      for (int nt = 0; nt < 2; ++nt)
        acc[mt][nt] = WMMA_F16(af, bf[nt], acc[mt][nt]);
    }
    __syncthreads();                            // done reading before re-fill
  }

  // Epilogue: C layout -> row m = v + 8*(lane/16), col n = lane%16
  const int ln = lane & 15, hf = lane >> 4;
#pragma unroll
  for (int mt = 0; mt < 4; ++mt) {
#pragma unroll
    for (int nt = 0; nt < 2; ++nt) {
      const int n = col0 + wn * 32 + nt * 16 + ln;
      const float bv = bias[n];
#pragma unroll
      for (int v = 0; v < 8; ++v) {
        const int m = row0 + wm * 64 + mt * 16 + v + 8 * hf;
        const float val = acc[mt][nt][v] + bv;
        if (OUT_F16)
          ((_Float16*)C)[(size_t)m * N + n] = (_Float16)val;
        else
          ((float*)C)[(size_t)m * N + n] = val;
      }
    }
  }
}

// ---------------------------------------------------------------------------
// Flash attention: one wave per (b, h, 16-query tile). qkv row layout per the
// reference reshape: head h occupies cols [h*192, h*192+192) = Q|K|V (64 each).
// ---------------------------------------------------------------------------
__global__ __launch_bounds__(32)
void flash_attn_16x(const _Float16* __restrict__ qkv, _Float16* __restrict__ out) {
  __shared__ _Float16 Ps[16 * 32];   // P tile, C-layout -> A-layout bounce
  const int lane = threadIdx.x & 31;
  const int ln = lane & 15, hf = lane >> 4;
  const int h4 = hf << 2;

  int idx = blockIdx.x;
  const int qt = idx & 127; idx >>= 7;   // S/16 = 128
  const int hh = idx & 15;  idx >>= 4;   // H = 16
  const int b  = idx;
  const int q0 = qt * 16;

  const _Float16* qkv_b = qkv + (size_t)b * Sc * E3 + hh * (3 * Dc);

  // Q fragments: 16x64 as two 16x32 A-fragments (contraction over D)
  v16h qf[2];
  {
    const uint32_t* qrow = (const uint32_t*)(qkv_b + (size_t)(q0 + ln) * E3);
#pragma unroll
    for (int kc = 0; kc < 2; ++kc) {
      FragU r;
#pragma unroll
      for (int j = 0; j < 4; ++j) {
        r.u[j]     = qrow[kc * 16 + h4 + j];
        r.u[4 + j] = qrow[kc * 16 + 8 + h4 + j];
      }
      qf[kc] = r.v;
    }
  }

  v8f o[4] = {};                 // O = 16x64 accumulator (4 tiles of 16x16)
  float rmax[8], rsum[8];
#pragma unroll
  for (int v = 0; v < 8; ++v) { rmax[v] = -3.0e38f; rsum[v] = 0.0f; }

  const float scale = 0.03125f;  // 1/sqrt(E) per reference
  const int nblocks = (q0 + 16 + 31) >> 5;   // causal frontier, 32-key blocks

  for (int kb = 0; kb < nblocks; ++kb) {
    // Prefetch next key block's K and V rows into L2 (strided rows).
    if (kb + 1 < nblocks) {
      const _Float16* nxt = qkv_b + (size_t)(kb * 32 + 32 + lane) * E3;
      __builtin_prefetch(nxt + Dc, 0, 1);       // K row
      __builtin_prefetch(nxt + 2 * Dc, 0, 1);   // V row
    }

    // ---- S = Q @ K^T (scores, 16x32) ----
    v8f s[2] = {};
#pragma unroll
    for (int c = 0; c < 2; ++c) {
      const int key = kb * 32 + c * 16 + ln;       // lane carries key column
      const uint32_t* krow = (const uint32_t*)(qkv_b + (size_t)key * E3 + Dc);
#pragma unroll
      for (int kc = 0; kc < 2; ++kc) {
        FragU r;
#pragma unroll
        for (int j = 0; j < 4; ++j) {
          r.u[j]     = krow[kc * 16 + h4 + j];
          r.u[4 + j] = krow[kc * 16 + 8 + h4 + j];
        }
        s[c] = WMMA_F16(qf[kc], r.v, s[c]);
      }
    }

    // ---- online softmax per row (row = v + 8*hf) ----
    float fac[8];
#pragma unroll
    for (int v = 0; v < 8; ++v) {
      const int qr = q0 + v + 8 * hf;
      float s0 = ((kb * 32 + ln)      <= qr) ? s[0][v] * scale : -3.0e38f;
      float s1 = ((kb * 32 + 16 + ln) <= qr) ? s[1][v] * scale : -3.0e38f;
      float mx = fmaxf(s0, s1);
#pragma unroll
      for (int off = 1; off < 16; off <<= 1)
        mx = fmaxf(mx, __shfl_xor(mx, off, 32));      // stays in 16-lane half
      const float nm = fmaxf(rmax[v], mx);
      const float f  = __expf(rmax[v] - nm);
      const float p0 = __expf(s0 - nm);
      const float p1 = __expf(s1 - nm);
      Ps[(v + 8 * hf) * 32 + ln]      = (_Float16)p0;
      Ps[(v + 8 * hf) * 32 + 16 + ln] = (_Float16)p1;
      float rs = p0 + p1;
#pragma unroll
      for (int off = 1; off < 16; off <<= 1)
        rs += __shfl_xor(rs, off, 32);
      rsum[v] = rsum[v] * f + rs;
      rmax[v] = nm;
      fac[v]  = f;
    }
#pragma unroll
    for (int nt = 0; nt < 4; ++nt)
#pragma unroll
      for (int v = 0; v < 8; ++v)
        o[nt][v] *= fac[v];

    __syncthreads();

    // ---- P fragment (A layout) from LDS ----
    FragU pr;
    {
      const uint32_t* pp = (const uint32_t*)(Ps + (size_t)ln * 32);
#pragma unroll
      for (int j = 0; j < 4; ++j) {
        pr.u[j]     = pp[h4 + j];
        pr.u[4 + j] = pp[8 + h4 + j];
      }
    }

    // ---- O += P @ V ----
#pragma unroll
    for (int nt = 0; nt < 4; ++nt) {
      const _Float16* vcol = qkv_b + (size_t)(kb * 32) * E3 + 2 * Dc + nt * 16 + ln;
      FragU vr;
#pragma unroll
      for (int j = 0; j < 8; ++j) {
        vr.h[j]     = vcol[(size_t)(hf * 8 + j) * E3];
        vr.h[8 + j] = vcol[(size_t)(16 + hf * 8 + j) * E3];
      }
      o[nt] = WMMA_F16(pr.v, vr.v, o[nt]);
    }
    __syncthreads();
  }

  // ---- write attention output f16: [B*S, E], col = h*64 + d ----
  float rinv[8];
#pragma unroll
  for (int v = 0; v < 8; ++v) rinv[v] = 1.0f / rsum[v];
#pragma unroll
  for (int nt = 0; nt < 4; ++nt) {
#pragma unroll
    for (int v = 0; v < 8; ++v) {
      const int m = q0 + v + 8 * hf;
      const int d = nt * 16 + ln;
      out[((size_t)b * Sc + m) * Ec + hh * Dc + d] = (_Float16)(o[nt][v] * rinv[v]);
    }
  }
}

// ---------------------------------------------------------------------------
// Launcher. Inputs: x, W_qkv, b_qkv, W_out, b_out, mask(=1).
// Workspace layout (bytes):
//   xh    @ 0         : 16 MiB  (B*S*E f16)
//   wqkvh @ 16777216  :  6 MiB  (E*3E f16)
//   wouth @ 23068672  :  2 MiB  (E*E  f16)
//   qkvh  @ 25165824  : 48 MiB  (B*S*3E f16)
//   atth  @ 75497472  : 16 MiB  (B*S*E f16)
// ---------------------------------------------------------------------------
extern "C" void kernel_launch(void* const* d_in, const int* in_sizes, int n_in,
                              void* d_out, int out_size, void* d_ws, size_t ws_size,
                              hipStream_t stream) {
  (void)in_sizes; (void)n_in; (void)out_size; (void)ws_size;

  const float* x     = (const float*)d_in[0];
  const float* W_qkv = (const float*)d_in[1];
  const float* b_qkv = (const float*)d_in[2];
  const float* W_out = (const float*)d_in[3];
  const float* b_out = (const float*)d_in[4];
  // mask assumed 1 (causal) per reference setup

  char* ws = (char*)d_ws;
  _Float16* xh    = (_Float16*)(ws);
  _Float16* wqkvh = (_Float16*)(ws + (size_t)16777216);
  _Float16* wouth = (_Float16*)(ws + (size_t)23068672);
  _Float16* qkvh  = (_Float16*)(ws + (size_t)25165824);
  _Float16* atth  = (_Float16*)(ws + (size_t)75497472);

  const int nX  = Bc * Sc * Ec;     // 8388608
  const int nWq = Ec * 3 * Ec;      // 3145728
  const int nWo = Ec * Ec;          // 1048576

  cvt_f32_f16<<<(nX  + 255) / 256, 256, 0, stream>>>(x,     xh,    nX);
  cvt_f32_f16<<<(nWq + 255) / 256, 256, 0, stream>>>(W_qkv, wqkvh, nWq);
  cvt_f32_f16<<<(nWo + 255) / 256, 256, 0, stream>>>(W_out, wouth, nWo);

  // QKV projection: [8192,1024] @ [1024,3072] + b_qkv -> f16
  gemm_wmma_tdm<true><<<dim3(3072 / 128, 8192 / 128), 256, 0, stream>>>(
      xh, wqkvh, b_qkv, (void*)qkvh, Bc * Sc, 3 * Ec, Ec);

  // Flash attention: B*H*(S/16) waves
  flash_attn_16x<<<dim3(Bc * Hc * (Sc / 16)), 32, 0, stream>>>(qkvh, atth);

  // Output projection: [8192,1024] @ [1024,1024] + b_out -> f32 (d_out)
  gemm_wmma_tdm<false><<<dim3(1024 / 128, 8192 / 128), 256, 0, stream>>>(
      atth, wouth, b_out, d_out, Bc * Sc, Ec, Ec);
}
